// GraphConvLayer_4715874091024
// MI455X (gfx1250) — matrix operand order repaired and checked by reading, not compiled, not run on hardware
//
#include <hip/hip_runtime.h>

typedef __attribute__((ext_vector_type(16))) __bf16 v16bf;
typedef __attribute__((ext_vector_type(8)))  __bf16 v8bf;
typedef __attribute__((ext_vector_type(8)))  float  v8f;

#define NODES   100000
#define IN_F    256
#define OUT_F   512
#define EDGES   300000
#define KTOT    512      // fused K: [x | agg]
#define KTILES  16       // KTOT / 32
#define MTILES  6250     // NODES / 16  (exact)
#define APAD    8        // LDS row pad (16B) -> rotates bank groups

// ---------------- helpers ----------------

__device__ __forceinline__ unsigned short f32_to_bf16_bits(float f) {
  union { float f; unsigned u; } v; v.f = f;
  unsigned u = v.u;
  unsigned r = u + 0x7FFFu + ((u >> 16) & 1u);   // round-to-nearest-even
  return (unsigned short)(r >> 16);
}

__device__ __forceinline__ void atomic_add_f32(float* p, float v) {
  __hip_atomic_fetch_add(p, v, __ATOMIC_RELAXED, __HIP_MEMORY_SCOPE_AGENT);
}

// ---------------- kernel 1: zero agg + counts (contiguous floats) ----------------

__global__ void zero_f32_kernel(float* __restrict__ p, long n) {
  long i = (long)blockIdx.x * blockDim.x + threadIdx.x;
  if (i < n) p[i] = 0.0f;
}

// ---------------- kernel 2: fp32 -> bf16 convert ----------------

__global__ void cvt_bf16_kernel(const float* __restrict__ src,
                                unsigned short* __restrict__ dst, long n) {
  long i = (long)blockIdx.x * blockDim.x + threadIdx.x;
  if (i < n) dst[i] = f32_to_bf16_bits(src[i]);
}

// ---------------- kernel 3: pack combined W into WMMA B-fragment order ----------------
// frag element j at wfrag[((nT*KTILES + kt)*32 + lane)*16 + j] holds
//   B[K = kt*32 + (j>>3)*16 + (lane>>4)*8 + (j&7)][col = nT*16 + (lane&15)]
// where B = [W_self ; W_neigh]  (512 x 512)

__global__ void pack_w_kernel(const float* __restrict__ Ws,
                              const float* __restrict__ Wn,
                              unsigned short* __restrict__ wfrag) {
  int tid = blockIdx.x * blockDim.x + threadIdx.x;   // 0 .. 262143
  int j    = tid & 15;
  int lane = (tid >> 4) & 31;
  int kt   = (tid >> 9) & 15;
  int nT   = tid >> 13;
  int K    = kt * 32 + (j >> 3) * 16 + (lane >> 4) * 8 + (j & 7);
  int col  = nT * 16 + (lane & 15);
  float v = (K < IN_F) ? Ws[(long)K * OUT_F + col]
                       : Wn[(long)(K - IN_F) * OUT_F + col];
  wfrag[tid] = f32_to_bf16_bits(v);
}

// ---------------- kernel 4: edge scatter (segment sum + degree) ----------------
// 64 threads per edge, float4 per thread (64*4 = 256 floats)

__global__ void scatter_kernel(const float* __restrict__ x,
                               const int* __restrict__ ei,   // [2, EDGES] int32
                               float* __restrict__ agg,
                               float* __restrict__ counts) {
  long gid = (long)blockIdx.x * blockDim.x + threadIdx.x;
  int edge = (int)(gid >> 6);
  int t    = (int)(gid & 63);
  if (edge >= EDGES) return;
  int s = ei[edge];
  int d = ei[EDGES + edge];
  const float4 v = ((const float4*)(x + (long)s * IN_F))[t];
  float* dst = agg + (long)d * IN_F + t * 4;
  atomic_add_f32(dst + 0, v.x);
  atomic_add_f32(dst + 1, v.y);
  atomic_add_f32(dst + 2, v.z);
  atomic_add_f32(dst + 3, v.w);
  if (t == 0) atomic_add_f32(counts + d, 1.0f);
}

// ---------------- kernel 5: mean-normalize + convert to bf16 ----------------

__global__ void normalize_kernel(const float* __restrict__ agg,
                                 const float* __restrict__ counts,
                                 unsigned short* __restrict__ aggb) {
  long i = (long)blockIdx.x * blockDim.x + threadIdx.x;
  if (i >= (long)NODES * IN_F) return;
  int node = (int)(i >> 8);                     // IN_F == 256
  float c = counts[node];
  c = (c < 1.0f) ? 1.0f : c;
  aggb[i] = f32_to_bf16_bits(agg[i] / c);
}

// ---------------- kernel 6: fused WMMA GEMM with LDS-staged A tile ----------------
// block = 8 waves; block covers rows [blockIdx.x*16, +16) with the fused K=512
// A tile staged once in LDS; wave w computes cols [w*64, +64) via 4 accumulators.

__global__ __launch_bounds__(256) void gemm_wmma_kernel(
    const __bf16* __restrict__ xb, const __bf16* __restrict__ aggb,
    const __bf16* __restrict__ wfrag,
    const float* __restrict__ bs, const float* __restrict__ bn,
    float* __restrict__ out) {
  __shared__ __align__(16) __bf16 As[16][KTOT + APAD];

  const int tid  = threadIdx.x;
  const int lane = tid & 31;
  const int wave = tid >> 5;
  const int l15  = lane & 15;
  const int half = lane >> 4;           // 0 or 1
  const long mBase = (long)blockIdx.x * 16;

  // cooperative stage of fused A tile: 16 rows x 512 bf16 (16 KB)
  // 1024 chunks of 8 bf16; 4 chunks per thread
  #pragma unroll
  for (int p = 0; p < 4; ++p) {
    const int cch = tid + p * 256;
    const int row = cch >> 6;           // 0..15
    const int k   = (cch & 63) * 8;     // 0..504
    const __bf16* src = (k < IN_F)
        ? (xb   + (mBase + row) * IN_F + k)
        : (aggb + (mBase + row) * IN_F + (k - IN_F));
    *(v8bf*)&As[row][k] = *(const v8bf*)src;
  }
  __syncthreads();

  v8f c[4] = {};

  // A-fragment per-lane base in LDS: row = l15, K offset half*8 (+16 for hi chunk)
  const __bf16* arow = &As[l15][half * 8];

  #pragma unroll
  for (int kt = 0; kt < KTILES; ++kt) {
    const int k0 = kt * 32;
    const v8bf lo = *(const v8bf*)(arow + k0);        // K = k0+half*8 + 0..7
    const v8bf hi = *(const v8bf*)(arow + k0 + 16);   // K = k0+half*8 + 16..23
    v16bf a;
    #pragma unroll
    for (int j = 0; j < 8; ++j) { a[j] = lo[j]; a[j + 8] = hi[j]; }

    #pragma unroll
    for (int i = 0; i < 4; ++i) {
      const int nT = wave * 4 + i;
      const v16bf b = *(const v16bf*)(wfrag + (((long)(nT * KTILES + kt) * 32 + lane) * 16));
      c[i] = __builtin_amdgcn_wmma_f32_16x16x32_bf16(
          /*neg_a=*/false, a, /*neg_b=*/false, b,
          /*c_mod=*/(short)0, c[i], /*reuse_a=*/false, /*reuse_b=*/false);
    }
  }

  // C/D layout: element g of v8f -> row mBase + g + half*8, col = nT*16 + l15
  #pragma unroll
  for (int i = 0; i < 4; ++i) {
    const int col = wave * 64 + i * 16 + l15;
    const float bias = bs[col] + bn[col];
    #pragma unroll
    for (int g = 0; g < 8; ++g) {
      const long row = mBase + g + half * 8;
      out[row * OUT_F + col] = c[i][g] + bias;
    }
  }
}

// ---------------- launcher ----------------

extern "C" void kernel_launch(void* const* d_in, const int* in_sizes, int n_in,
                              void* d_out, int out_size, void* d_ws, size_t ws_size,
                              hipStream_t stream) {
  const float* x      = (const float*)d_in[0];
  const int*   ei     = (const int*)d_in[1];     // edge_index (int32 under default JAX x64-off)
  const float* Wself  = (const float*)d_in[2];
  const float* bself  = (const float*)d_in[3];
  const float* Wneigh = (const float*)d_in[4];
  const float* bneigh = (const float*)d_in[5];
  float* out = (float*)d_out;

  // workspace layout (16B-aligned sections)
  float* agg            = (float*)d_ws;                          // NODES*IN_F f32
  float* counts         = agg + (long)NODES * IN_F;              // NODES f32
  unsigned short* xb    = (unsigned short*)(counts + NODES);     // NODES*IN_F bf16
  unsigned short* aggb  = xb + (long)NODES * IN_F;               // NODES*IN_F bf16
  unsigned short* wfrag = aggb + (long)NODES * IN_F;             // KTOT*OUT_F bf16

  const long nAgg  = (long)NODES * IN_F + NODES;   // agg + counts contiguous
  const long nFeat = (long)NODES * IN_F;

  // 1. zero agg + counts
  zero_f32_kernel<<<(int)((nAgg + 255) / 256), 256, 0, stream>>>(agg, nAgg);
  // 2. x -> bf16
  cvt_bf16_kernel<<<(int)((nFeat + 255) / 256), 256, 0, stream>>>(x, xb, nFeat);
  // 3. pack W fragments (512*512 elements)
  pack_w_kernel<<<(KTOT * OUT_F) / 256, 256, 0, stream>>>(Wself, Wneigh, wfrag);
  // 4. edge scatter (64 threads/edge)
  scatter_kernel<<<(int)(((long)EDGES * 64 + 255) / 256), 256, 0, stream>>>(x, ei, agg, counts);
  // 5. normalize + bf16
  normalize_kernel<<<(int)((nFeat + 255) / 256), 256, 0, stream>>>(agg, counts, aggb);
  // 6. fused WMMA GEMM (LDS-staged A)
  gemm_wmma_kernel<<<MTILES, 256, 0, stream>>>((const __bf16*)xb, (const __bf16*)aggb,
                                               (const __bf16*)wfrag, bself, bneigh, out);
}
